// GDAModule_55937654063134
// MI455X (gfx1250) — matrix operand order repaired and checked by reading, not compiled, not run on hardware
//
#include <hip/hip_runtime.h>

// ---------------------------------------------------------------------------
// Types for CDNA5 WMMA (gfx1250, wave32)
// ---------------------------------------------------------------------------
typedef __bf16 bf16_t;
typedef __attribute__((ext_vector_type(16))) __bf16 v16bf;
typedef __attribute__((ext_vector_type(8)))  float  v8f;

#define C_DIM   64
#define D_IN    128
#define D_ST    16
#define DT_R    4
#define BN      2
#define HH      128
#define WW      128
#define H2      64
#define W2      64
#define L2D     4096            // H2*W2
#define NPIX    (BN*HH*WW)      // 32768
#define NTOK    (NPIX*2)        // 65536 temporal tokens
#define OUTSZ   (BN*C_DIM*HH*WW) // 2097152

__device__ __forceinline__ float siluf(float x)     { return x / (1.f + __expf(-x)); }
__device__ __forceinline__ float sigmf(float x)     { return 1.f / (1.f + __expf(-x)); }
__device__ __forceinline__ float softplusf(float x) { return (x > 20.f) ? x : log1pf(__expf(x)); }

// ---------------------------------------------------------------------------
// Generic f32 -> bf16 conversion
// ---------------------------------------------------------------------------
__global__ void k_cvt_bf16(const float* __restrict__ in, bf16_t* __restrict__ out, int n) {
    int i = blockIdx.x * blockDim.x + threadIdx.x;
    if (i < n) out[i] = (bf16_t)in[i];
}

// ---------------------------------------------------------------------------
// WMMA bf16 GEMM:  Y[M x N] (f32) = A[M x K] (bf16, row major) * B[N x K]^T
// One wave computes a 16 x (NT*16) strip: the A fragment is loaded once per
// K-step and reused for NT consecutive v_wmma_f32_16x16x32_bf16 ops.
// Out-of-range N columns are handled by clamping the B row (loads stay in
// bounds) and skipping the store -- no EXEC manipulation in the hot loop.
// M must be a multiple of 16, K a multiple of 32.
// ---------------------------------------------------------------------------
template <int NT>
__global__ void __launch_bounds__(256)
k_gemm_bf16_nt(const bf16_t* __restrict__ A, const bf16_t* __restrict__ B,
               float* __restrict__ Y, int M, int N, int K) {
    const int wave = threadIdx.x >> 5;
    const int lane = threadIdx.x & 31;
    const int ntN  = (N + 15) >> 4;
    const int ntM  = M >> 4;
    const int ngrp = (ntN + NT - 1) / NT;
    const int tile = blockIdx.x * (blockDim.x >> 5) + wave;
    if (tile >= ntM * ngrp) return;             // wave-uniform: EXEC stays full
    const int mt = tile / ngrp, g = tile % ngrp;
    const int half = lane >> 4, lr = lane & 15;
    const int n0base = g * NT * 16;

    // A lane layout (16x32 bf16): half=0 -> K {0..7, 16..23}; half=1 -> +8
    const int m = (mt << 4) + lr;
    const bf16_t* arow = A + (size_t)m * K + (half << 3);

    // B lane layout (32x16 bf16): half=0 -> K {0..7, 16..23}; half=1 -> +8
    const bf16_t* brow[NT];
    #pragma unroll
    for (int j = 0; j < NT; ++j) {
        int n = n0base + j * 16 + lr;
        int nc = (n < N) ? n : (N - 1);         // clamp: load valid, store never
        brow[j] = B + (size_t)nc * K + (half << 3);
    }

    v8f acc[NT];
    #pragma unroll
    for (int j = 0; j < NT; ++j) acc[j] = {0.f, 0.f, 0.f, 0.f, 0.f, 0.f, 0.f, 0.f};

    for (int kk = 0; kk < K; kk += 32) {
        union { uint4 q[2]; v16bf v; } ua;
        ua.q[0] = *(const uint4*)(arow + kk);
        ua.q[1] = *(const uint4*)(arow + kk + 16);
        #pragma unroll
        for (int j = 0; j < NT; ++j) {
            union { uint4 q[2]; v16bf v; } ub;
            ub.q[0] = *(const uint4*)(brow[j] + kk);
            ub.q[1] = *(const uint4*)(brow[j] + kk + 16);
            acc[j] = __builtin_amdgcn_wmma_f32_16x16x32_bf16(
                         false, ua.v, false, ub.v, (short)0, acc[j], false, false);
        }
    }
    const int mbase = (mt << 4) + (half << 3);  // D: lanes>=16 hold M rows 8..15
    #pragma unroll
    for (int j = 0; j < NT; ++j) {
        int n = n0base + j * 16 + lr;
        if (n < N) {
            #pragma unroll
            for (int r = 0; r < 8; ++r)
                Y[(size_t)(mbase + r) * N + n] = acc[j][r];
        }
    }
}

// ---------------------------------------------------------------------------
// Patch embed: conv k=2 s=2 (C->C) + bias, output token-major (b, p, c)
// ---------------------------------------------------------------------------
__global__ void k_patch_embed(const float* __restrict__ x, const float* __restrict__ w,
                              const float* __restrict__ bias, float* __restrict__ out) {
    int id = blockIdx.x * blockDim.x + threadIdx.x;
    if (id >= BN * L2D * C_DIM) return;
    int c = id & 63, p = (id >> 6) & (L2D - 1), b = id >> 18;
    int h2 = p >> 6, w2 = p & 63;
    const float* xb = x + (size_t)b * C_DIM * (HH * WW);
    const float* wc = w + c * (C_DIM * 4);
    float acc = bias[c];
    for (int ci = 0; ci < C_DIM; ++ci) {
        const float* xp = xb + (size_t)ci * (HH * WW) + (2 * h2) * WW + 2 * w2;
        const float* wk = wc + ci * 4;
        acc += wk[0] * xp[0] + wk[1] * xp[1] + wk[2] * xp[WW] + wk[3] * xp[WW + 1];
    }
    out[id] = acc;
}

// GroupNorm stats: one block per (b, group); 4 groups of 16 channels
__global__ void k_gn_stats(const float* __restrict__ xp, float* __restrict__ stat) {
    int bg = blockIdx.x, b = bg >> 2, g = bg & 3;
    float s = 0.f, ss = 0.f;
    for (int i = threadIdx.x; i < 16 * L2D; i += blockDim.x) {
        int c = g * 16 + (i & 15), p = i >> 4;
        float v = xp[((size_t)(b << 12) + p) * C_DIM + c];
        s += v; ss += v * v;
    }
    __shared__ float rs[256], rq[256];
    rs[threadIdx.x] = s; rq[threadIdx.x] = ss;
    __syncthreads();
    for (int st = 128; st > 0; st >>= 1) {
        if (threadIdx.x < st) { rs[threadIdx.x] += rs[threadIdx.x + st]; rq[threadIdx.x] += rq[threadIdx.x + st]; }
        __syncthreads();
    }
    if (threadIdx.x == 0) {
        float m = rs[0] * (1.f / 65536.f);
        float v = rq[0] * (1.f / 65536.f) - m * m;
        stat[bg * 2] = m;
        stat[bg * 2 + 1] = rsqrtf(v + 1e-5f);
    }
}

__global__ void k_gn_apply_silu(const float* __restrict__ xp, const float* __restrict__ stat,
                                const float* __restrict__ g, const float* __restrict__ be,
                                float* __restrict__ out) {
    int id = blockIdx.x * blockDim.x + threadIdx.x;
    if (id >= BN * L2D * C_DIM) return;
    int c = id & 63, b = id >> 18;
    int gi = b * 4 + (c >> 4);
    out[id] = siluf((xp[id] - stat[gi * 2]) * stat[gi * 2 + 1] * g[c] + be[c]);
}

// Token LayerNorm (f32 in, bf16 out), Cd <= 128
__global__ void k_ln_bf16(const float* __restrict__ in, const float* __restrict__ g,
                          const float* __restrict__ be, bf16_t* __restrict__ out,
                          int T, int Cd) {
    int t = blockIdx.x * blockDim.x + threadIdx.x;
    if (t >= T) return;
    const float* row = in + (size_t)t * Cd;
    float s = 0.f, ss = 0.f;
    for (int c = 0; c < Cd; ++c) { float v = row[c]; s += v; ss += v * v; }
    float m = s / Cd, r = rsqrtf(ss / Cd - m * m + 1e-5f);
    bf16_t* o = out + (size_t)t * Cd;
    for (int c = 0; c < Cd; ++c) o[c] = (bf16_t)((row[c] - m) * r * g[c] + be[c]);
}

// Depthwise 3x3 conv + bias + SiLU over xi half of xz tokens -> (b, d, l) layout
__global__ void k_dwconv3_silu(const float* __restrict__ xz, const float* __restrict__ w,
                               const float* __restrict__ bias, float* __restrict__ out) {
    int id = blockIdx.x * blockDim.x + threadIdx.x;
    if (id >= BN * D_IN * L2D) return;
    int l = id & (L2D - 1), d = (id >> 12) & 127, b = id >> 19;
    int h = l >> 6, wd = l & 63;
    const float* wk = w + d * 9;
    float acc = bias[d];
    for (int kh = 0; kh < 3; ++kh) {
        int hh = h + kh - 1; if (hh < 0 || hh > 63) continue;
        for (int kw = 0; kw < 3; ++kw) {
            int wc = wd + kw - 1; if (wc < 0 || wc > 63) continue;
            acc += wk[kh * 3 + kw] * xz[((size_t)(b * L2D + hh * 64 + wc)) * 256 + d];
        }
    }
    out[id] = siluf(acc);
}

// Build 4-direction token streams xs (k,b,l,d) in bf16
__global__ void k_make_xs(const float* __restrict__ xic, bf16_t* __restrict__ out) {
    int id = blockIdx.x * blockDim.x + threadIdx.x;
    if (id >= 4 * BN * L2D * D_IN) return;
    int d = id & 127, l = (id >> 7) & (L2D - 1), b = (id >> 19) & 1, k = id >> 20;
    int lsrc;
    if (k == 0)      lsrc = l;
    else if (k == 1) lsrc = (l & 63) * 64 + (l >> 6);
    else if (k == 2) lsrc = (L2D - 1) - l;
    else { int lr = (L2D - 1) - l; lsrc = (lr & 63) * 64 + (lr >> 6); }
    out[id] = (bf16_t)xic[((size_t)(b * D_IN + d)) * L2D + lsrc];
}

// dt projection + softplus for SS2D -> dts (k,b,l,d)
__global__ void k_dt_s(const float* __restrict__ dbl, const float* __restrict__ dtw,
                       const float* __restrict__ dtb, float* __restrict__ dts) {
    int id = blockIdx.x * blockDim.x + threadIdx.x;
    if (id >= 4 * BN * L2D * D_IN) return;
    int d = id & 127, l = (id >> 7) & (L2D - 1), b = (id >> 19) & 1, k = id >> 20;
    const float* dr = dbl + (size_t)(k * 8192 + b * L2D + l) * 36;
    const float* dw = dtw + ((size_t)k * D_IN + d) * 4;
    float a = dtb[k * D_IN + d] + dw[0] * dr[0] + dw[1] * dr[1] + dw[2] * dr[2] + dw[3] * dr[3];
    dts[id] = softplusf(a);
}

// Sequential selective scan, L=4096. One block per (k,b); thread = d.
__global__ void __launch_bounds__(128)
k_scan_s(const float* __restrict__ dbl, const float* __restrict__ dts,
         const bf16_t* __restrict__ xs, const float* __restrict__ A_log,
         const float* __restrict__ Dp, float* __restrict__ ys) {
    int kb = blockIdx.x, k = kb >> 1, b = kb & 1;
    int d = threadIdx.x;
    float A[D_ST], h[D_ST];
    #pragma unroll
    for (int s = 0; s < D_ST; ++s) {
        A[s] = -__expf(A_log[((size_t)k * D_IN + d) * D_ST + s]);
        h[s] = 0.f;
    }
    float Dd = Dp[k * D_IN + d];
    __shared__ float bc[32];
    size_t base = (size_t)kb * L2D;
    size_t drow = (size_t)(k * 8192 + b * L2D) * 36;
    for (int l = 0; l < L2D; ++l) {
        __syncthreads();
        if (threadIdx.x < 32) bc[threadIdx.x] = dbl[drow + (size_t)l * 36 + 4 + threadIdx.x];
        if (threadIdx.x == 0 && l + 8 < L2D)
            __builtin_prefetch(dbl + drow + (size_t)(l + 8) * 36, 0, 0);
        __syncthreads();
        float dt = dts[(base + l) * D_IN + d];
        float u  = (float)xs[(base + l) * D_IN + d];
        float du = dt * u, y = 0.f;
        #pragma unroll
        for (int s = 0; s < D_ST; ++s) {
            h[s] = __expf(dt * A[s]) * h[s] + du * bc[s];
            y += h[s] * bc[16 + s];
        }
        ys[(base + l) * D_IN + d] = y + u * Dd;
    }
}

#define YSIDX(k, b, li, d) (((((size_t)((k) * 2 + (b))) << 12) + (li)) * D_IN + (d))

// Combine 4 directions + LayerNorm(d=128) + gate with silu(z) -> bf16 tokens
__global__ void __launch_bounds__(256)
k_combine_s(const float* __restrict__ ys, const float* __restrict__ xz_s,
            const float* __restrict__ g, const float* __restrict__ be,
            bf16_t* __restrict__ out) {
    int wid = (blockIdx.x * blockDim.x + threadIdx.x) >> 5;
    int lane = threadIdx.x & 31;
    if (wid >= BN * L2D) return;
    int b = wid >> 12, l = wid & (L2D - 1);
    int h = l >> 6, w = l & 63;
    int lt = w * 64 + h;                       // transpose index
    float y[4], s = 0.f, ss = 0.f;
    #pragma unroll
    for (int j = 0; j < 4; ++j) {
        int d = lane + 32 * j;
        float v = ys[YSIDX(0, b, l, d)]
                + ys[YSIDX(2, b, (L2D - 1) - l, d)]
                + ys[YSIDX(1, b, lt, d)]
                + ys[YSIDX(3, b, (L2D - 1) - lt, d)];
        y[j] = v; s += v; ss += v * v;
    }
    for (int o = 16; o; o >>= 1) { s += __shfl_xor(s, o); ss += __shfl_xor(ss, o); }
    float m = s * (1.f / 128.f), r = rsqrtf(ss * (1.f / 128.f) - m * m + 1e-5f);
    #pragma unroll
    for (int j = 0; j < 4; ++j) {
        int d = lane + 32 * j;
        float z = xz_s[((size_t)wid) * 256 + 128 + d];
        out[(size_t)wid * D_IN + d] = (bf16_t)(((y[j] - m) * r * g[d] + be[d]) * siluf(z));
    }
}

// Residual add + transpose tokens -> (b, c, p)
__global__ void k_residual_chw(const float* __restrict__ xin_tok, const float* __restrict__ yout,
                               float* __restrict__ xv) {
    int id = blockIdx.x * blockDim.x + threadIdx.x;
    if (id >= BN * C_DIM * L2D) return;
    int p = id & (L2D - 1), c = (id >> 12) & 63, b = id >> 18;
    size_t tok = (size_t)b * L2D + p;
    xv[id] = xin_tok[tok * C_DIM + c] + yout[tok * C_DIM + c];
}

// Bilinear x2 upsample (half-pixel centers), (b,c,64,64) -> (b,c,128,128)
__global__ void k_upsample2(const float* __restrict__ xv, float* __restrict__ xsp) {
    int id = blockIdx.x * blockDim.x + threadIdx.x;
    if (id >= OUTSZ) return;
    int w = id & 127, h = (id >> 7) & 127, c = (id >> 14) & 63, b = id >> 20;
    float fy = 0.5f * h - 0.25f, fx = 0.5f * w - 0.25f;
    int y0 = (int)floorf(fy), x0 = (int)floorf(fx);
    float wy = fy - y0, wx = fx - x0;
    int y0c = min(63, max(0, y0)), y1c = min(63, max(0, y0 + 1));
    int x0c = min(63, max(0, x0)), x1c = min(63, max(0, x0 + 1));
    const float* p = xv + ((size_t)(b * C_DIM + c)) * L2D;
    float v = (1.f - wy) * ((1.f - wx) * p[y0c * 64 + x0c] + wx * p[y0c * 64 + x1c])
            + wy        * ((1.f - wx) * p[y1c * 64 + x0c] + wx * p[y1c * 64 + x1c]);
    xsp[id] = v;
}

// Temporal branch: build (token, c) bf16 sequence from x1/x2
__global__ void k_make_seq(const float* __restrict__ x1, const float* __restrict__ x2,
                           bf16_t* __restrict__ out) {
    int id = blockIdx.x * blockDim.x + threadIdx.x;
    if (id >= NTOK * C_DIM) return;
    int c = id & 63, tok = id >> 6;
    int t = tok & 1, pix = tok >> 1;
    int b = pix >> 14, hw = pix & 16383;
    const float* src = t ? x2 : x1;
    out[id] = (bf16_t)src[((size_t)(b * C_DIM + c)) * (HH * WW) + hw];
}

// Fused per-pixel L=2 Mamba: conv1d + x_proj + dt + scan + gate + out_proj +
// LN + SiLU + residual.  One wave per pixel, LDS for cross-d dots.
__global__ void __launch_bounds__(256)
k_spe_fused(const float* __restrict__ xz, const float* __restrict__ x1,
            const float* __restrict__ x2, const float* __restrict__ conv_w,
            const float* __restrict__ conv_b, const float* __restrict__ xproj_w,
            const float* __restrict__ dt_w, const float* __restrict__ dt_b,
            const float* __restrict__ A_log, const float* __restrict__ Dp,
            const float* __restrict__ out_w, const float* __restrict__ ln_g,
            const float* __restrict__ ln_b, float* __restrict__ out1,
            float* __restrict__ out2) {
    __shared__ float l_xc[8][2][D_IN];
    __shared__ float l_dbl[8][2][36];
    __shared__ float l_yg[8][2][D_IN];
    int wv = threadIdx.x >> 5, lane = threadIdx.x & 31;
    int pix = blockIdx.x * 8 + wv;               // 4096 blocks * 8 waves = 32768
    size_t t0 = (size_t)pix * 2 * 256, t1 = t0 + 256;

    float xc[2][4], zz[2][4];
    #pragma unroll
    for (int j = 0; j < 4; ++j) {
        int d = lane + 32 * j;
        float xi0 = xz[t0 + d], xi1 = xz[t1 + d];
        zz[0][j] = xz[t0 + 128 + d];
        zz[1][j] = xz[t1 + 128 + d];
        float w2c = conv_w[d * 4 + 2], w3c = conv_w[d * 4 + 3], cb = conv_b[d];
        xc[0][j] = siluf(xi0 * w3c + cb);                 // causal k=4 conv, L=2
        xc[1][j] = siluf(xi0 * w2c + xi1 * w3c + cb);
        l_xc[wv][0][d] = xc[0][j];
        l_xc[wv][1][d] = xc[1][j];
    }
    __syncthreads();
    for (int cc = lane; cc < 36; cc += 32) {             // x_proj (36 x 128)
        const float* wr = xproj_w + cc * D_IN;
        float a0 = 0.f, a1 = 0.f;
        for (int d = 0; d < D_IN; ++d) {
            float w = wr[d];
            a0 += w * l_xc[wv][0][d];
            a1 += w * l_xc[wv][1][d];
        }
        l_dbl[wv][0][cc] = a0;
        l_dbl[wv][1][cc] = a1;
    }
    __syncthreads();
    float y[2][4];
    #pragma unroll
    for (int j = 0; j < 4; ++j) {
        int d = lane + 32 * j;
        const float* dw = dt_w + d * 4;
        float dt0 = dt_b[d], dt1 = dt_b[d];
        #pragma unroll
        for (int r = 0; r < 4; ++r) {
            dt0 += dw[r] * l_dbl[wv][0][r];
            dt1 += dw[r] * l_dbl[wv][1][r];
        }
        dt0 = softplusf(dt0); dt1 = softplusf(dt1);
        float u0 = xc[0][j], u1 = xc[1][j], Dd = Dp[d];
        float y0 = 0.f, y1 = 0.f;
        const float* Ar = A_log + d * D_ST;
        #pragma unroll
        for (int s = 0; s < D_ST; ++s) {
            float A = -__expf(Ar[s]);
            float h = dt0 * u0 * l_dbl[wv][0][4 + s];    // h0 = 0
            y0 += h * l_dbl[wv][0][20 + s];
            h = __expf(dt1 * A) * h + dt1 * u1 * l_dbl[wv][1][4 + s];
            y1 += h * l_dbl[wv][1][20 + s];
        }
        y[0][j] = (y0 + u0 * Dd) * siluf(zz[0][j]);
        y[1][j] = (y1 + u1 * Dd) * siluf(zz[1][j]);
    }
    __syncthreads();
    #pragma unroll
    for (int j = 0; j < 4; ++j) {
        int d = lane + 32 * j;
        l_yg[wv][0][d] = y[0][j];
        l_yg[wv][1][d] = y[1][j];
    }
    __syncthreads();
    float ym[2][2];
    #pragma unroll
    for (int i = 0; i < 2; ++i) {                        // out_proj (64 x 128)
        int cc = lane + 32 * i;
        const float* wr = out_w + cc * D_IN;
        float a0 = 0.f, a1 = 0.f;
        for (int d = 0; d < D_IN; ++d) {
            float w = wr[d];
            a0 += w * l_yg[wv][0][d];
            a1 += w * l_yg[wv][1][d];
        }
        ym[0][i] = a0; ym[1][i] = a1;
    }
    int b = pix >> 14, hw = pix & 16383;
    #pragma unroll
    for (int t = 0; t < 2; ++t) {                        // LN(64) + SiLU + residual
        float s = ym[t][0] + ym[t][1];
        float ss = ym[t][0] * ym[t][0] + ym[t][1] * ym[t][1];
        for (int o = 16; o; o >>= 1) { s += __shfl_xor(s, o); ss += __shfl_xor(ss, o); }
        float m = s * (1.f / 64.f), r = rsqrtf(ss * (1.f / 64.f) - m * m + 1e-5f);
        float* dst = t ? out2 : out1;
        const float* src = t ? x2 : x1;
        #pragma unroll
        for (int i = 0; i < 2; ++i) {
            int cc = lane + 32 * i;
            float yp = siluf((ym[t][i] - m) * r * ln_g[cc] + ln_b[cc]);
            size_t gi = ((size_t)(b * C_DIM + cc)) * (HH * WW) + hw;
            dst[gi] = src[gi] + yp;
        }
    }
}

// Adaptive gated fusion: per-pixel channel dots -> sigmoid gates -> blend
__global__ void __launch_bounds__(256)
k_fusion(const float* __restrict__ xsp, const float* __restrict__ x1o,
         const float* __restrict__ x2o, const float* __restrict__ gAw,
         const float* __restrict__ gAb, const float* __restrict__ gBw,
         const float* __restrict__ gBb, float* __restrict__ out) {
    int wv = threadIdx.x >> 5, lane = threadIdx.x & 31;
    int pix = blockIdx.x * 8 + wv;
    int b = pix >> 14, hw = pix & 16383;
    float sa = 0.f, sb = 0.f, xs[2], xd[2];
    #pragma unroll
    for (int i = 0; i < 2; ++i) {
        int c = lane + 32 * i;
        size_t gi = ((size_t)(b * C_DIM + c)) * (HH * WW) + hw;
        xs[i] = xsp[gi];
        xd[i] = x2o[gi] - x1o[gi];
        sa += xs[i] * gAw[c];
        sb += xd[i] * gBw[c];
    }
    for (int o = 16; o; o >>= 1) { sa += __shfl_xor(sa, o); sb += __shfl_xor(sb, o); }
    float gA = sigmf(sa + gAb[0]), gB = sigmf(sb + gBb[0]);
    #pragma unroll
    for (int i = 0; i < 2; ++i) {
        int c = lane + 32 * i;
        size_t gi = ((size_t)(b * C_DIM + c)) * (HH * WW) + hw;
        out[gi] = gA * xs[i] + gB * xd[i];
    }
}

// ---------------------------------------------------------------------------
// Host-side launch
// ---------------------------------------------------------------------------
extern "C" void kernel_launch(void* const* d_in, const int* in_sizes, int n_in,
                              void* d_out, int out_size, void* d_ws, size_t ws_size,
                              hipStream_t stream) {
    (void)in_sizes; (void)n_in; (void)out_size; (void)ws_size;
    const float* x_spa     = (const float*)d_in[0];
    const float* x1        = (const float*)d_in[1];
    const float* x2        = (const float*)d_in[2];
    const float* m_in_w    = (const float*)d_in[3];
    const float* m_conv_w  = (const float*)d_in[4];
    const float* m_conv_b  = (const float*)d_in[5];
    const float* m_xproj_w = (const float*)d_in[6];
    const float* m_dt_w    = (const float*)d_in[7];
    const float* m_dt_b    = (const float*)d_in[8];
    const float* m_A_log   = (const float*)d_in[9];
    const float* m_D       = (const float*)d_in[10];
    const float* m_out_w   = (const float*)d_in[11];
    const float* spe_ln_g  = (const float*)d_in[12];
    const float* spe_ln_b  = (const float*)d_in[13];
    const float* pe_w      = (const float*)d_in[14];
    const float* pe_b      = (const float*)d_in[15];
    const float* gn_g      = (const float*)d_in[16];
    const float* gn_b      = (const float*)d_in[17];
    const float* vss_ln_g  = (const float*)d_in[18];
    const float* vss_ln_b  = (const float*)d_in[19];
    const float* s_in_w    = (const float*)d_in[20];
    const float* s_conv_w  = (const float*)d_in[21];
    const float* s_conv_b  = (const float*)d_in[22];
    const float* s_xproj_w = (const float*)d_in[23];
    const float* s_dt_w    = (const float*)d_in[24];
    const float* s_dt_b    = (const float*)d_in[25];
    const float* s_A_log   = (const float*)d_in[26];
    const float* s_D       = (const float*)d_in[27];
    const float* s_oln_g   = (const float*)d_in[28];
    const float* s_oln_b   = (const float*)d_in[29];
    const float* s_out_w   = (const float*)d_in[30];
    const float* gA_w      = (const float*)d_in[31];
    const float* gA_b      = (const float*)d_in[32];
    const float* gB_w      = (const float*)d_in[33];
    const float* gB_b      = (const float*)d_in[34];

    char* ws = (char*)d_ws;
    size_t off = 0;
    auto alloc = [&](size_t bytes) -> size_t {
        size_t o = off; off = (off + bytes + 255) & ~(size_t)255; return o;
    };
    // persistent across both phases
    size_t o_wsin  = alloc((size_t)256 * 64 * 2);        // s_in_w bf16
    size_t o_wsxp  = alloc((size_t)4 * 36 * 128 * 2);    // s_xproj_w bf16
    size_t o_wsout = alloc((size_t)64 * 128 * 2);        // s_out_w bf16
    size_t o_wmin  = alloc((size_t)256 * 64 * 2);        // m_in_w bf16
    size_t o_xsp   = alloc((size_t)OUTSZ * 4);           // upsampled spatial branch
    size_t SCR = off;

    // ---- phase A scratch (SS2D / spatial branch) ----
    size_t o_xp   = alloc((size_t)BN * L2D * C_DIM * 4);
    size_t o_stat = alloc(16 * 4);
    size_t o_xin  = alloc((size_t)BN * L2D * C_DIM * 4);
    size_t o_lnbf = alloc((size_t)BN * L2D * C_DIM * 2);
    size_t o_xzs  = alloc((size_t)BN * L2D * 256 * 4);
    size_t o_xic  = alloc((size_t)BN * D_IN * L2D * 4);
    size_t o_xsbf = alloc((size_t)4 * BN * L2D * D_IN * 2);
    size_t o_dbl  = alloc((size_t)4 * BN * L2D * 36 * 4);
    size_t o_dts  = alloc((size_t)4 * BN * L2D * D_IN * 4);
    size_t o_ys   = alloc((size_t)4 * BN * L2D * D_IN * 4);
    size_t o_ygbf = alloc((size_t)BN * L2D * D_IN * 2);
    size_t o_yout = alloc((size_t)BN * L2D * C_DIM * 4);
    size_t o_xv   = alloc((size_t)BN * C_DIM * L2D * 4);

    // ---- phase B scratch (temporal branch) overlaps phase A (stream-serial) ----
    off = SCR;
    size_t o_seq  = alloc((size_t)NTOK * C_DIM * 2);
    size_t o_xz2  = alloc((size_t)NTOK * 256 * 4);

    float* out0 = (float*)d_out;
    float* out1 = out0 + OUTSZ;
    float* out2 = out0 + 2 * (size_t)OUTSZ;

    // GEMM dispatcher: NT=4 strips when 4 | ntN, NT=3 for N=36
    auto launch_gemm = [&](const bf16_t* A, const bf16_t* B, float* Y,
                           int M, int N, int K) {
        int ntM = M >> 4, ntN = (N + 15) >> 4;
        if ((ntN & 3) == 0) {
            int waves = ntM * (ntN >> 2);
            k_gemm_bf16_nt<4><<<(waves + 7) / 8, 256, 0, stream>>>(A, B, Y, M, N, K);
        } else {
            int waves = ntM * ((ntN + 2) / 3);
            k_gemm_bf16_nt<3><<<(waves + 7) / 8, 256, 0, stream>>>(A, B, Y, M, N, K);
        }
    };

    // Weight conversions to bf16
    k_cvt_bf16<<<(16384 + 255) / 256, 256, 0, stream>>>(s_in_w,    (bf16_t*)(ws + o_wsin), 16384);
    k_cvt_bf16<<<(18432 + 255) / 256, 256, 0, stream>>>(s_xproj_w, (bf16_t*)(ws + o_wsxp), 18432);
    k_cvt_bf16<<<(8192  + 255) / 256, 256, 0, stream>>>(s_out_w,   (bf16_t*)(ws + o_wsout), 8192);
    k_cvt_bf16<<<(16384 + 255) / 256, 256, 0, stream>>>(m_in_w,    (bf16_t*)(ws + o_wmin), 16384);

    // ======== Phase A: spatial branch ========
    k_patch_embed<<<2048, 256, 0, stream>>>(x_spa, pe_w, pe_b, (float*)(ws + o_xp));
    k_gn_stats<<<8, 256, 0, stream>>>((float*)(ws + o_xp), (float*)(ws + o_stat));
    k_gn_apply_silu<<<2048, 256, 0, stream>>>((float*)(ws + o_xp), (float*)(ws + o_stat),
                                              gn_g, gn_b, (float*)(ws + o_xin));
    k_ln_bf16<<<32, 256, 0, stream>>>((float*)(ws + o_xin), vss_ln_g, vss_ln_b,
                                      (bf16_t*)(ws + o_lnbf), BN * L2D, C_DIM);
    // SS2D in_proj: (8192,64) x (256,64)^T  -> WMMA
    launch_gemm((bf16_t*)(ws + o_lnbf), (bf16_t*)(ws + o_wsin),
                (float*)(ws + o_xzs), 8192, 256, 64);
    k_dwconv3_silu<<<4096, 256, 0, stream>>>((float*)(ws + o_xzs), s_conv_w, s_conv_b,
                                             (float*)(ws + o_xic));
    k_make_xs<<<16384, 256, 0, stream>>>((float*)(ws + o_xic), (bf16_t*)(ws + o_xsbf));
    for (int k = 0; k < 4; ++k)     // x_proj per direction: (8192,128)x(36,128)^T -> WMMA
        launch_gemm((bf16_t*)(ws + o_xsbf) + (size_t)k * 8192 * 128,
                    (bf16_t*)(ws + o_wsxp) + (size_t)k * 36 * 128,
                    (float*)(ws + o_dbl) + (size_t)k * 8192 * 36, 8192, 36, 128);
    k_dt_s<<<16384, 256, 0, stream>>>((float*)(ws + o_dbl), s_dt_w, s_dt_b,
                                      (float*)(ws + o_dts));
    k_scan_s<<<8, 128, 0, stream>>>((float*)(ws + o_dbl), (float*)(ws + o_dts),
                                    (bf16_t*)(ws + o_xsbf), s_A_log, s_D,
                                    (float*)(ws + o_ys));
    k_combine_s<<<1024, 256, 0, stream>>>((float*)(ws + o_ys), (float*)(ws + o_xzs),
                                          s_oln_g, s_oln_b, (bf16_t*)(ws + o_ygbf));
    // SS2D out_proj: (8192,128)x(64,128)^T -> WMMA
    launch_gemm((bf16_t*)(ws + o_ygbf), (bf16_t*)(ws + o_wsout),
                (float*)(ws + o_yout), 8192, 64, 128);
    k_residual_chw<<<2048, 256, 0, stream>>>((float*)(ws + o_xin), (float*)(ws + o_yout),
                                             (float*)(ws + o_xv));
    k_upsample2<<<8192, 256, 0, stream>>>((float*)(ws + o_xv), (float*)(ws + o_xsp));

    // ======== Phase B: temporal branch ========
    k_make_seq<<<16384, 256, 0, stream>>>(x1, x2, (bf16_t*)(ws + o_seq));
    // spe in_proj: (65536,64)x(256,64)^T -> WMMA (largest GEMM, 2.1 GFLOP)
    launch_gemm((bf16_t*)(ws + o_seq), (bf16_t*)(ws + o_wmin),
                (float*)(ws + o_xz2), 65536, 256, 64);
    k_spe_fused<<<4096, 256, 0, stream>>>((float*)(ws + o_xz2), x1, x2,
                                          m_conv_w, m_conv_b, m_xproj_w, m_dt_w, m_dt_b,
                                          m_A_log, m_D, m_out_w, spe_ln_g, spe_ln_b,
                                          out1, out2);

    // ======== Fusion ========
    k_fusion<<<4096, 256, 0, stream>>>((float*)(ws + o_xsp), out1, out2,
                                       gA_w, gA_b, gB_w, gB_b, out0);
}
